// ContrastiveHead_myself_4535485464678
// MI455X (gfx1250) — compile-verified, matrix operand order checked by reading, not run on hardware
//
#include <hip/hip_runtime.h>

// ---------------------------------------------------------------------------
// MI455X (gfx1250) implementation of the contrastive head.
//   - bf16 NHWC activations (channels contiguous) -> A fragments are two
//     aligned 16B global loads per lane, matching the CDNA5 16-bit A layout.
//   - weights device-prepacked into B-fragment lane order, staged into LDS
//     with gfx1250 async global->LDS DMA (ASYNCcnt tracked).
//   - conv3x3 as implicit GEMM: 72 x v_wmma_f32_16x16x32_bf16 per wave.
//   - BN stats via global f32 atomics, finalize + fused apply/ReLU (b128 I/O).
// ---------------------------------------------------------------------------

typedef __bf16 bf16_t;
typedef __bf16 v16bf __attribute__((ext_vector_type(16)));
typedef __bf16 v8bf  __attribute__((ext_vector_type(8)));
typedef float  v8f   __attribute__((ext_vector_type(8)));

// Problem constants
static constexpr int kN   = 4;
static constexpr int kC   = 64;
static constexpr int kH   = 256;
static constexpr int kW   = 256;
static constexpr int kHW  = kH * kW;           // 65536
static constexpr int kNHW = kN * kHW;          // 262144
static constexpr int kS   = 500;

// ---------------------------------------------------------------------------
// gfx1250 async global->LDS copy helpers (inline asm; ASYNCcnt tracked)
// ---------------------------------------------------------------------------
__device__ __forceinline__ void async_copy_b128(unsigned lds_byte_off, const void* gptr) {
  asm volatile("global_load_async_to_lds_b128 %0, %1, off"
               :: "v"(lds_byte_off), "v"((unsigned long long)(uintptr_t)gptr)
               : "memory");
}
__device__ __forceinline__ void wait_async0() {
#if __has_builtin(__builtin_amdgcn_s_wait_asynccnt)
  __builtin_amdgcn_s_wait_asynccnt(0);
#else
  asm volatile("s_wait_asynccnt 0" ::: "memory");
#endif
}

// ---------------------------------------------------------------------------
// Pack fp32 NCHW input -> bf16 NHWC (float4 reads; 4 pixels per thread)
// ---------------------------------------------------------------------------
__global__ void k_pack_input(const float* __restrict__ x, bf16_t* __restrict__ dst) {
  size_t i4 = ((size_t)blockIdx.x * 256 + threadIdx.x) * 4;   // NCHW flat, over 16,777,216
  float4 v = *(const float4*)(x + i4);
  int n = (int)(i4 >> 22);
  int c = (int)((i4 >> 16) & 63);
  int p = (int)(i4 & 65535);                                  // p is a multiple of 4
  size_t base = ((size_t)((n << 16) | p)) * 64 + c;
  dst[base]       = (bf16_t)v.x;
  dst[base + 64]  = (bf16_t)v.y;
  dst[base + 128] = (bf16_t)v.z;
  dst[base + 192] = (bf16_t)v.w;
}

// ---------------------------------------------------------------------------
// Pack conv weights [O=64][I=64][3][3] fp32 -> bf16 B-fragments:
// layout [tap(9)][kc(2)][nt(4)][lane(32)][elem(16)]
// elem e of lane l holds B[K][Ncol] with
//   K = kc*32 + (l<16?0:8) + (e&7) + ((e>>3)<<4),  Ncol = l&15, oc = nt*16+Ncol
// ---------------------------------------------------------------------------
__global__ void k_pack_conv_w(const float* __restrict__ w, bf16_t* __restrict__ frag) {
  int i = blockIdx.x * 256 + threadIdx.x;      // 36864 total
  int e    =  i        & 15;
  int lane = (i >> 4)  & 31;
  int nt   = (i >> 9)  & 3;
  int kc   = (i >> 11) & 1;
  int tap  =  i >> 12;                         // 0..8
  int K  = kc * 32 + ((lane < 16) ? 0 : 8) + (e & 7) + ((e >> 3) << 4);
  int oc = nt * 16 + (lane & 15);
  int ky = tap / 3, kx = tap % 3;
  frag[i] = (bf16_t)w[((oc * 64 + K) * 3 + ky) * 3 + kx];
}

__global__ void k_pack_linear_w(const float* __restrict__ w, bf16_t* __restrict__ frag) {
  int i = blockIdx.x * 256 + threadIdx.x;      // 4096 total
  int e    =  i        & 15;
  int lane = (i >> 4)  & 31;
  int nt   = (i >> 9)  & 3;
  int kc   = (i >> 11) & 1;
  int K  = kc * 32 + ((lane < 16) ? 0 : 8) + (e & 7) + ((e >> 3) << 4);
  int oc = nt * 16 + (lane & 15);
  frag[i] = (bf16_t)w[oc * 64 + K];            // B[K=in][N=out] = W[out][in]
}

__global__ void k_zero(float* p, int n) {
  int i = blockIdx.x * 256 + threadIdx.x;
  if (i < n) p[i] = 0.f;
}

// ---------------------------------------------------------------------------
// Implicit-GEMM 3x3 conv.  Wave = 16 pixels x 64 oc, 72 WMMA / wave.
// Writes pre-BN bf16 NHWC and accumulates per-channel sum / sumsq.
// ---------------------------------------------------------------------------
__global__ void __launch_bounds__(256) k_conv_wmma(
    const bf16_t* __restrict__ in, const bf16_t* __restrict__ wfrag,
    const float* __restrict__ bias, bf16_t* __restrict__ out,
    float* __restrict__ gsum, float* __restrict__ gsq) {
  extern __shared__ bf16_t sW[];               // 36864 bf16 = 72 KB
  const int tid = threadIdx.x;
  {                                            // async DMA B fragments into LDS
    const char* src = (const char*)wfrag;
    for (int j = tid; j < 73728 / 16; j += 256)
      async_copy_b128((unsigned)(j * 16), src + (size_t)j * 16);
    wait_async0();
  }
  __syncthreads();

  const int wv   = tid >> 5;
  const int lane = tid & 31;
  const int tileBase = blockIdx.x * 128 + wv * 16;   // 16 pixels per wave
  const int mrow = lane & 15;
  const int pix  = tileBase + mrow;
  const int n = pix >> 16, h = (pix >> 8) & 255, w = pix & 255;
  const int koff = (lane < 16) ? 0 : 8;

  v8f acc0 = {}, acc1 = {}, acc2 = {}, acc3 = {};

#pragma unroll
  for (int tap = 0; tap < 9; ++tap) {
    const int hh = h + tap / 3 - 1;
    const int ww = w + tap % 3 - 1;
    const bool valid = ((unsigned)hh < 256u) && ((unsigned)ww < 256u);
    const size_t qbase = ((size_t)((n << 16) | ((hh & 255) << 8) | (ww & 255))) * 64;
    if (valid) __builtin_prefetch(in + qbase + 128 * 64, 0, 0);  // global_prefetch_b8
#pragma unroll
    for (int kc = 0; kc < 2; ++kc) {
      v16bf a = {};
      if (valid) {
        const bf16_t* ap = in + qbase + kc * 32 + koff;
        v8bf lo = *(const v8bf*)ap;            // K  0..7 (+8 for hi lanes)
        v8bf hi = *(const v8bf*)(ap + 16);     // K 16..23 (+8 for hi lanes)
        a = __builtin_shufflevector(lo, hi, 0,1,2,3,4,5,6,7,8,9,10,11,12,13,14,15);
      }
      const bf16_t* bp = sW + (size_t)((tap * 2 + kc) * 4) * 512 + lane * 16;
      v16bf b0 = *(const v16bf*)(bp);
      v16bf b1 = *(const v16bf*)(bp + 512);
      v16bf b2 = *(const v16bf*)(bp + 1024);
      v16bf b3 = *(const v16bf*)(bp + 1536);
      acc0 = __builtin_amdgcn_wmma_f32_16x16x32_bf16(false, a, false, b0, (short)0, acc0, false, false);
      acc1 = __builtin_amdgcn_wmma_f32_16x16x32_bf16(false, a, false, b1, (short)0, acc1, false, false);
      acc2 = __builtin_amdgcn_wmma_f32_16x16x32_bf16(false, a, false, b2, (short)0, acc2, false, false);
      acc3 = __builtin_amdgcn_wmma_f32_16x16x32_bf16(false, a, false, b3, (short)0, acc3, false, false);
    }
  }

  // Epilogue: C/D layout -> lane gives N, vgpr r gives M (lanes>=16: M=r+8)
  const int ncol = lane & 15;
  const int mhi  = (lane < 16) ? 0 : 8;
  v8f accs[4] = {acc0, acc1, acc2, acc3};
#pragma unroll
  for (int nt = 0; nt < 4; ++nt) {
    const int oc = nt * 16 + ncol;
    const float bv = bias[oc];
    float s = 0.f, s2 = 0.f;
#pragma unroll
    for (int r = 0; r < 8; ++r) {
      float v = accs[nt][r] + bv;
      out[(size_t)(tileBase + mhi + r) * 64 + oc] = (bf16_t)v;
      s += v; s2 += v * v;
    }
    atomicAdd(&gsum[oc], s);
    atomicAdd(&gsq[oc], s2);
  }
}

// ---------------------------------------------------------------------------
// Linear layer (1x1): wave = 16 rows x 64 oc, 8 WMMA / wave.
// ---------------------------------------------------------------------------
__global__ void __launch_bounds__(256) k_linear_wmma(
    const bf16_t* __restrict__ in, const bf16_t* __restrict__ wfrag,
    const float* __restrict__ bias, bf16_t* __restrict__ out,
    float* __restrict__ gsum, float* __restrict__ gsq) {
  extern __shared__ bf16_t sW[];               // 4096 bf16 = 8 KB
  const int tid = threadIdx.x;
  {
    const char* src = (const char*)wfrag;
    for (int j = tid; j < 8192 / 16; j += 256)
      async_copy_b128((unsigned)(j * 16), src + (size_t)j * 16);
    wait_async0();
  }
  __syncthreads();

  const int wv   = tid >> 5;
  const int lane = tid & 31;
  const int tileBase = blockIdx.x * 128 + wv * 16;
  const int row  = tileBase + (lane & 15);
  const int koff = (lane < 16) ? 0 : 8;

  v8f acc0 = {}, acc1 = {}, acc2 = {}, acc3 = {};
#pragma unroll
  for (int kc = 0; kc < 2; ++kc) {
    const bf16_t* ap = in + (size_t)row * 64 + kc * 32 + koff;
    v8bf lo = *(const v8bf*)ap;
    v8bf hi = *(const v8bf*)(ap + 16);
    v16bf a = __builtin_shufflevector(lo, hi, 0,1,2,3,4,5,6,7,8,9,10,11,12,13,14,15);
    const bf16_t* bp = sW + (size_t)(kc * 4) * 512 + lane * 16;
    v16bf b0 = *(const v16bf*)(bp);
    v16bf b1 = *(const v16bf*)(bp + 512);
    v16bf b2 = *(const v16bf*)(bp + 1024);
    v16bf b3 = *(const v16bf*)(bp + 1536);
    acc0 = __builtin_amdgcn_wmma_f32_16x16x32_bf16(false, a, false, b0, (short)0, acc0, false, false);
    acc1 = __builtin_amdgcn_wmma_f32_16x16x32_bf16(false, a, false, b1, (short)0, acc1, false, false);
    acc2 = __builtin_amdgcn_wmma_f32_16x16x32_bf16(false, a, false, b2, (short)0, acc2, false, false);
    acc3 = __builtin_amdgcn_wmma_f32_16x16x32_bf16(false, a, false, b3, (short)0, acc3, false, false);
  }

  const int ncol = lane & 15;
  const int mhi  = (lane < 16) ? 0 : 8;
  v8f accs[4] = {acc0, acc1, acc2, acc3};
#pragma unroll
  for (int nt = 0; nt < 4; ++nt) {
    const int oc = nt * 16 + ncol;
    const float bv = bias[oc];
    float s = 0.f, s2 = 0.f;
#pragma unroll
    for (int r = 0; r < 8; ++r) {
      float v = accs[nt][r] + bv;
      out[(size_t)(tileBase + mhi + r) * 64 + oc] = (bf16_t)v;
      s += v; s2 += v * v;
    }
    atomicAdd(&gsum[oc], s);
    atomicAdd(&gsq[oc], s2);
  }
}

// ---------------------------------------------------------------------------
// BN finalize: per-channel scale/shift from accumulated sum/sumsq.
// ---------------------------------------------------------------------------
__global__ void k_bn_finalize(const float* __restrict__ gsum, const float* __restrict__ gsq,
                              const float* __restrict__ g, const float* __restrict__ be,
                              float* __restrict__ scale, float* __restrict__ shift) {
  int c = threadIdx.x;
  if (c < 64) {
    const float inv = 1.0f / (float)kNHW;
    float m  = gsum[c] * inv;
    float v  = gsq[c] * inv - m * m;
    float sc = g[c] * rsqrtf(v + 1e-5f);
    scale[c] = sc;
    shift[c] = be[c] - m * sc;
  }
}

// BN apply + ReLU, 8 channels per thread (b128 load/store);
// optionally also writes fp32 NCHW (x_enc output).
__global__ void k_bn_apply(const bf16_t* __restrict__ pre, const float* __restrict__ scale,
                           const float* __restrict__ shift, bf16_t* __restrict__ outbf,
                           float* __restrict__ nchw) {
  size_t base = ((size_t)blockIdx.x * 256 + threadIdx.x) * 8;   // over NHW*64
  int c0 = (int)(base & 63);
  v8bf pv = *(const v8bf*)(pre + base);
  v8bf ov = {};
  float rv[8];
#pragma unroll
  for (int j = 0; j < 8; ++j) {
    int c = c0 + j;
    float v = (float)pv[j];
    float r = fmaxf(scale[c] * v + shift[c], 0.0f);
    ov[j] = (bf16_t)r;
    rv[j] = r;
  }
  *(v8bf*)(outbf + base) = ov;
  if (nchw) {
    size_t row = base >> 6;              // n*HW + p
    size_t n = row >> 16;
    size_t p = row & 65535;
#pragma unroll
    for (int j = 0; j < 8; ++j)
      nchw[((n * 64 + (size_t)(c0 + j)) << 16) + p] = rv[j];
  }
}

// ---------------------------------------------------------------------------
// Masked mean pools: block per image; groups of 64 lanes own one channel each.
// ---------------------------------------------------------------------------
__global__ void __launch_bounds__(256) k_query(const bf16_t* __restrict__ y,
    const unsigned char* __restrict__ posm, const unsigned char* __restrict__ negm,
    float* __restrict__ qpos, float* __restrict__ qneg) {
  __shared__ float sP[4][64], sN[4][64], sCp[4], sCn[4];
  const int n = blockIdx.x;
  const int g = threadIdx.x >> 6, c = threadIdx.x & 63;
  float ps = 0.f, ns = 0.f, cp = 0.f, cn = 0.f;
  for (int p = g; p < kHW; p += 4) {
    const int row = (n << 16) + p;
    const float f = (float)y[(size_t)row * 64 + c];
    const unsigned char pm = posm[row], nm = negm[row];
    if (pm) ps += f;
    if (nm) ns += f;
    if (c == 0) { cp += pm ? 1.f : 0.f; cn += nm ? 1.f : 0.f; }
  }
  sP[g][c] = ps; sN[g][c] = ns;
  if (c == 0) { sCp[g] = cp; sCn[g] = cn; }
  __syncthreads();
  if (g == 0) {
    float tp  = sP[0][c] + sP[1][c] + sP[2][c] + sP[3][c];
    float tn  = sN[0][c] + sN[1][c] + sN[2][c] + sN[3][c];
    float ccp = sCp[0] + sCp[1] + sCp[2] + sCp[3];
    float ccn = sCn[0] + sCn[1] + sCn[2] + sCn[3];
    qpos[n * 64 + c] = tp / ccp;
    qneg[n * 64 + c] = tn / ccn;
  }
}

// Gather: 8 channels per thread (b128 reads of the source row segment)
__global__ void k_gather(const bf16_t* __restrict__ y, const int* __restrict__ idx,
                         float* __restrict__ out) {
  int t = blockIdx.x * 256 + threadIdx.x;      // 2000 rows * 8 segs = 16000
  if (t >= kN * kS * 8) return;
  int r  = t >> 3;
  int c8 = (t & 7) * 8;
  v8bf v = *(const v8bf*)(y + (size_t)idx[r] * 64 + c8);
#pragma unroll
  for (int j = 0; j < 8; ++j)
    out[(size_t)r * 64 + c8 + j] = (float)v[j];
}

// ---------------------------------------------------------------------------
// Orchestration
// ---------------------------------------------------------------------------
extern "C" void kernel_launch(void* const* d_in, const int* in_sizes, int n_in,
                              void* d_out, int out_size, void* d_ws, size_t ws_size,
                              hipStream_t stream) {
  (void)in_sizes; (void)n_in; (void)out_size; (void)ws_size;
  const float* x = (const float*)d_in[0];
  const unsigned char* posm = (const unsigned char*)d_in[1];
  const unsigned char* negm = (const unsigned char*)d_in[2];
  const int* idxEP = (const int*)d_in[3];
  const int* idxEN = (const int*)d_in[4];
  const int* idxHP = (const int*)d_in[5];
  const int* idxHN = (const int*)d_in[6];
  // layers: e_w1(7..10), e_w2(11..14), p_w1(15..18), p_w2(19..22)
  const float* cw[4]  = {(const float*)d_in[7],  (const float*)d_in[11], (const float*)d_in[15], (const float*)d_in[19]};
  const float* cb[4]  = {(const float*)d_in[8],  (const float*)d_in[12], (const float*)d_in[16], (const float*)d_in[20]};
  const float* cg[4]  = {(const float*)d_in[9],  (const float*)d_in[13], (const float*)d_in[17], (const float*)d_in[21]};
  const float* cbe[4] = {(const float*)d_in[10], (const float*)d_in[14], (const float*)d_in[18], (const float*)d_in[22]};
  const float* lw  = (const float*)d_in[23];
  const float* lb  = (const float*)d_in[24];
  const float* lg  = (const float*)d_in[25];
  const float* lbe = (const float*)d_in[26];

  // workspace layout
  const size_t actBytes = (size_t)kNHW * 64 * 2;      // 33,554,432
  char* ws = (char*)d_ws;
  bf16_t* actA = (bf16_t*)(ws);
  bf16_t* actB = (bf16_t*)(ws + actBytes);
  bf16_t* actP = (bf16_t*)(ws + 2 * actBytes);
  bf16_t* wf[4];
  for (int i = 0; i < 4; ++i) wf[i] = (bf16_t*)(ws + 3 * actBytes + (size_t)i * 73728);
  bf16_t* wfl   = (bf16_t*)(ws + 3 * actBytes + 4 * 73728);
  float*  stats = (float*)(ws + 3 * actBytes + 4 * 73728 + 8192);
  float *gsum = stats, *gsq = stats + 64, *gscale = stats + 128, *gshift = stats + 192;

  // output layout (tuple concatenated)
  float* out    = (float*)d_out;
  float* o_xenc = out;                          // [4,64,256,256]
  float* o_qpos = out + 16777216;               // [4,1,64]
  float* o_qneg = o_qpos + 256;                 // [4,1,64]
  float* o_sep  = o_qneg + 256;                 // [2000,64]
  float* o_sen  = o_sep + 128000;
  float* o_shp  = o_sen + 128000;
  float* o_shn  = o_shp + 128000;

  // pack input + weights
  k_pack_input<<<16384, 256, 0, stream>>>(x, actA);
  for (int i = 0; i < 4; ++i) k_pack_conv_w<<<144, 256, 0, stream>>>(cw[i], wf[i]);
  k_pack_linear_w<<<16, 256, 0, stream>>>(lw, wfl);

  // 4 conv + BN + ReLU layers (ping-pong actA/actB; actP = pre-BN)
  bf16_t* cin = actA;
  bf16_t* cout = actB;
  for (int layer = 0; layer < 4; ++layer) {
    k_zero<<<1, 128, 0, stream>>>(stats, 128);
    k_conv_wmma<<<2048, 256, 73728, stream>>>(cin, wf[layer], cb[layer], actP, gsum, gsq);
    k_bn_finalize<<<1, 64, 0, stream>>>(gsum, gsq, cg[layer], cbe[layer], gscale, gshift);
    k_bn_apply<<<8192, 256, 0, stream>>>(actP, gscale, gshift, cout,
                                         (layer == 1) ? o_xenc : (float*)nullptr);
    bf16_t* t = cin; cin = cout; cout = t;
  }

  // linear + BN1d + ReLU -> y
  k_zero<<<1, 128, 0, stream>>>(stats, 128);
  k_linear_wmma<<<2048, 256, 8192, stream>>>(cin, wfl, lb, actP, gsum, gsq);
  k_bn_finalize<<<1, 64, 0, stream>>>(gsum, gsq, lg, lbe, gscale, gshift);
  k_bn_apply<<<8192, 256, 0, stream>>>(actP, gscale, gshift, cout, (float*)nullptr);
  bf16_t* yb = cout;

  // pooled queries + gathers
  k_query<<<4, 256, 0, stream>>>(yb, posm, negm, o_qpos, o_qneg);
  k_gather<<<63, 256, 0, stream>>>(yb, idxEP, o_sep);
  k_gather<<<63, 256, 0, stream>>>(yb, idxEN, o_sen);
  k_gather<<<63, 256, 0, stream>>>(yb, idxHP, o_shp);
  k_gather<<<63, 256, 0, stream>>>(yb, idxHN, o_shn);
}
